// HAN6Model_56495999812299
// MI455X (gfx1250) — matrix operand order repaired and checked by reading, not compiled, not run on hardware
//
#include <hip/hip_runtime.h>
#include <math.h>

#define NP_ 100000
#define NO_ 200000
#define E_  500000
#define IN_DIM_ 512
#define D_ 256

typedef __attribute__((ext_vector_type(16))) __bf16        v16bf;
typedef __attribute__((ext_vector_type(8)))  float         v8f;
typedef __attribute__((ext_vector_type(8)))  unsigned int  v8u;

// ---------- helpers ----------
__device__ __forceinline__ unsigned short f2bf(float f) {
    union { float f; unsigned u; } c; c.f = f;
    unsigned r = c.u + 0x7FFFu + ((c.u >> 16) & 1u);   // round-to-nearest-even
    return (unsigned short)(r >> 16);
}
// monotone float<->uint encoding for atomicMax on floats
__device__ __forceinline__ unsigned encf(float f) {
    union { float f; unsigned u; } c; c.f = f;
    return (c.u & 0x80000000u) ? ~c.u : (c.u | 0x80000000u);
}
__device__ __forceinline__ float decf(unsigned u) {
    union { unsigned u; float f; } c;
    c.u = (u & 0x80000000u) ? (u & 0x7FFFFFFFu) : ~u;
    return c.f;
}
#define ENC_NEG_INF 0x007FFFFFu   // encf(-inf)

// ---------- prep: convert + transpose Wm (K-major -> N-major, bf16) ----------
__global__ void prep_weights(const float* __restrict__ Wm_p, const float* __restrict__ Wm_o,
                             unsigned short* __restrict__ WmTp, unsigned short* __restrict__ WmTo) {
    int i = blockIdx.x * 256 + threadIdx.x;
    if (i < IN_DIM_ * D_) {
        int k = i >> 8, n = i & 255;           // Wm row-major [512,256]
        WmTp[n * IN_DIM_ + k] = f2bf(Wm_p[i]); // WmT [256,512] K-contiguous
        WmTo[n * IN_DIM_ + k] = f2bf(Wm_o[i]);
    }
}

// ---------- prep: fold W, bW, a_w into matvec vectors ----------
__global__ void prep_uvec(const float* __restrict__ W, const float* __restrict__ bW,
                          const float* __restrict__ a_w,
                          float* __restrict__ u_src, float* __restrict__ u_dst,
                          float* __restrict__ consts) {
    int k = threadIdx.x;
    float s0 = 0.f, s1 = 0.f;
    for (int n = 0; n < D_; ++n) {
        float w = W[k * D_ + n];
        s0 += w * a_w[n];
        s1 += w * a_w[D_ + n];
    }
    u_src[k] = s0;
    u_dst[k] = s1;
    if (k == 0) {
        float c0 = 0.f, c1 = 0.f;
        for (int n = 0; n < D_; ++n) { c0 += bW[n] * a_w[n]; c1 += bW[n] * a_w[D_ + n]; }
        consts[0] = c0; consts[1] = c1;
    }
}

// ---------- init: segment-max / denom / agg region ----------
__global__ void init_bufs(unsigned* __restrict__ mmax, float* __restrict__ denom,
                          float* __restrict__ agg_out, long long total_out) {
    long long i = (long long)blockIdx.x * 256 + threadIdx.x;
    long long stride = (long long)gridDim.x * 256;
    for (long long j = i; j < total_out; j += stride) agg_out[j] = 0.f;
    for (long long j = i; j < NP_; j += stride) { mmax[j] = ENC_NEG_INF; denom[j] = 0.f; }
}

// ---------- fused feature mapping: relu(X@Wm+b) via bf16 WMMA ----------
// 8 waves/block; block owns 16 rows, each wave owns a 16-col slice.
// Also computes s[row] = h . u + const, and (OTHER) normalized ho + row norms.
template <int OTHER>
__global__ __launch_bounds__(256)
void map_gemm(const float* __restrict__ X, const unsigned short* __restrict__ WmT,
              const float* __restrict__ bias, const float* __restrict__ uvec,
              const float* __restrict__ consts, int cidx,
              float* __restrict__ s_out, float* __restrict__ out2, float* __restrict__ rn) {
    __shared__ unsigned short ldsA[16 * IN_DIM_];   // 16 KB bf16 A tile
    __shared__ float lds_s[16];
    __shared__ float lds_ss[16];
    __shared__ float lds_inv[16];

    const int tid  = threadIdx.x;
    const int lane = tid & 31;
    const int wave = tid >> 5;
    const int row0 = blockIdx.x * 16;

    // stage A tile: fp32 -> bf16 into LDS (read features exactly once)
    for (int i = tid; i < 16 * IN_DIM_; i += 256) {
        int r = i >> 9, k = i & (IN_DIM_ - 1);
        ldsA[i] = f2bf(X[(long long)(row0 + r) * IN_DIM_ + k]);
    }
    if (tid < 16) { lds_s[tid] = 0.f; lds_ss[tid] = 0.f; }
    __syncthreads();

    // lane layouts (wave32): A and B both use contiguous k = 16*(lane/16) + s
    const int nloc   = lane & 15;
    const int col    = wave * 16 + nloc;            // output column this lane owns
    const int m      = lane & 15;                   // A row this lane feeds
    const int khalfA = (lane < 16) ? 0 : 16;        // A: 16x32 bf16
    const int khalfB = (lane < 16) ? 0 : 16;        // B: 32x16 bf16

    v8f c = {};
    const unsigned short* Bcol = WmT + (long long)col * IN_DIM_;

    for (int t = 0; t < IN_DIM_ / 32; ++t) {
        const unsigned short* Arow = &ldsA[m * IN_DIM_ + t * 32];
        v8u au, bu;
#pragma unroll
        for (int vg = 0; vg < 8; ++vg) {            // A: paired K elements per VGPR
            int kin = khalfA + vg * 2;
            au[vg] = *(const unsigned int*)(Arow + kin);
        }
        const unsigned short* Bp = Bcol + t * 32 + khalfB;
#pragma unroll
        for (int vg = 0; vg < 8; ++vg) {            // B: 16 contiguous bf16 along K
            bu[vg] = *(const unsigned int*)(Bp + vg * 2);
        }
        v16bf A = __builtin_bit_cast(v16bf, au);
        v16bf B = __builtin_bit_cast(v16bf, bu);
        // (neg_a, A, neg_b, B, c_mod, C, reuse_a, reuse_b)
        c = __builtin_amdgcn_wmma_f32_16x16x32_bf16(false, A, false, B, (short)0, c,
                                                    false, false);
    }

    // epilogue: bias + relu, score matvec, (OTHER) row sum-of-squares
    const float bcol = bias[col];
    const float ucol = uvec[col];
    float h[8];
#pragma unroll
    for (int v = 0; v < 8; ++v) {                   // C layout: row = v + 8*(lane>=16)
        float hv = fmaxf(c[v] + bcol, 0.f);
        h[v] = hv;
        float p = hv * ucol;                        // reduce across 16 cols of this wave
        p += __shfl_xor(p, 1); p += __shfl_xor(p, 2);
        p += __shfl_xor(p, 4); p += __shfl_xor(p, 8);
        if (nloc == 0) atomicAdd(&lds_s[v + ((lane >> 4) << 3)], p);
        if (OTHER) {
            float q = hv * hv;
            q += __shfl_xor(q, 1); q += __shfl_xor(q, 2);
            q += __shfl_xor(q, 4); q += __shfl_xor(q, 8);
            if (nloc == 0) atomicAdd(&lds_ss[v + ((lane >> 4) << 3)], q);
        }
    }
    __syncthreads();
    if (tid < 16) {
        s_out[row0 + tid] = lds_s[tid] + consts[cidx];
        if (OTHER) {
            float nrm = fmaxf(sqrtf(lds_ss[tid]), 1e-12f);
            rn[row0 + tid]  = nrm;
            lds_inv[tid]    = 1.0f / nrm;
        }
    }
    if (OTHER) {
        __syncthreads();
#pragma unroll
        for (int v = 0; v < 8; ++v) {
            int row = v + ((lane >> 4) << 3);
            out2[(long long)(row0 + row) * D_ + col] = h[v] * lds_inv[row];
        }
    }
}

// ---------- edge phase ----------
__global__ void edge_scores(const int* __restrict__ src, const int* __restrict__ dst,
                            const float* __restrict__ s_src, const float* __restrict__ s_dst,
                            const float* __restrict__ a_b,
                            float* __restrict__ ebuf, unsigned* __restrict__ mmax) {
    int i = blockIdx.x * 256 + threadIdx.x;
    if (i >= E_) return;
    float e = s_src[src[i]] + s_dst[dst[i]] + a_b[0];
    ebuf[i] = e;
    atomicMax(&mmax[src[i]], encf(e));
}

__global__ void edge_exp(const int* __restrict__ src, const unsigned* __restrict__ mmax,
                         float* __restrict__ ebuf, float* __restrict__ denom) {
    int i = blockIdx.x * 256 + threadIdx.x;
    if (i >= E_) return;
    int s = src[i];
    float ex = __expf(ebuf[i] - decf(mmax[s]));
    ebuf[i] = ex;
    atomicAdd(&denom[s], ex);
}

// one wave per edge: scatter alpha * ho[dst] (= out2*rownorm) into agg rows
__global__ __launch_bounds__(256)
void edge_agg(const int* __restrict__ src, const int* __restrict__ dst,
              const float* __restrict__ ebuf, const float* __restrict__ denom,
              const float* __restrict__ rn, const float* __restrict__ out2,
              float* __restrict__ out1) {
    int lane = threadIdx.x & 31;
    int eidx = blockIdx.x * 8 + (threadIdx.x >> 5);
    if (eidx >= E_) return;
    int s = src[eidx], dd = dst[eidx];
    float scale = (ebuf[eidx] / denom[s]) * rn[dd];
    const float* hrow = out2 + (long long)dd * D_;
    float*       arow = out1 + (long long)s  * D_;
#pragma unroll
    for (int c0 = 0; c0 < D_; c0 += 32)
        atomicAdd(&arow[c0 + lane], hrow[c0 + lane] * scale);
}

// ---------- final in-place L2 normalize of agg (one wave per row) ----------
__global__ __launch_bounds__(256)
void final_norm(float* __restrict__ out1) {
    int lane = threadIdx.x & 31;
    int row  = blockIdx.x * 8 + (threadIdx.x >> 5);
    if (row >= NP_) return;
    float* p = out1 + (long long)row * D_;
    float v[8], ss = 0.f;
#pragma unroll
    for (int j = 0; j < 8; ++j) { v[j] = p[lane + 32 * j]; ss += v[j] * v[j]; }
    ss += __shfl_xor(ss, 16); ss += __shfl_xor(ss, 8); ss += __shfl_xor(ss, 4);
    ss += __shfl_xor(ss, 2);  ss += __shfl_xor(ss, 1);
    float inv = 1.0f / fmaxf(sqrtf(ss), 1e-12f);
#pragma unroll
    for (int j = 0; j < 8; ++j) p[lane + 32 * j] = v[j] * inv;
}

// ---------- launch ----------
extern "C" void kernel_launch(void* const* d_in, const int* in_sizes, int n_in,
                              void* d_out, int out_size, void* d_ws, size_t ws_size,
                              hipStream_t stream) {
    const float* feat_prefix = (const float*)d_in[0];
    const float* feat_other  = (const float*)d_in[1];
    const int*   src         = (const int*)  d_in[2];
    const int*   dst         = (const int*)  d_in[3];
    const float* Wm_p        = (const float*)d_in[4];
    const float* bm_p        = (const float*)d_in[5];
    const float* Wm_o        = (const float*)d_in[6];
    const float* bm_o        = (const float*)d_in[7];
    const float* W           = (const float*)d_in[8];
    const float* bW          = (const float*)d_in[9];
    const float* a_w         = (const float*)d_in[10];
    const float* a_b         = (const float*)d_in[11];

    float* out  = (float*)d_out;
    float* out1 = out;                       // l2norm(agg)  [NP, D]
    float* out2 = out + (size_t)NP_ * D_;    // l2norm(ho)   [NO, D]

    char* ws = (char*)d_ws;
    unsigned short* WmTp = (unsigned short*)ws; ws += (size_t)IN_DIM_ * D_ * 2;
    unsigned short* WmTo = (unsigned short*)ws; ws += (size_t)IN_DIM_ * D_ * 2;
    float* u_src  = (float*)ws;    ws += D_ * 4;
    float* u_dst  = (float*)ws;    ws += D_ * 4;
    float* consts = (float*)ws;    ws += 16 * 4;
    float* s_src  = (float*)ws;    ws += (size_t)NP_ * 4;
    float* s_dst  = (float*)ws;    ws += (size_t)NO_ * 4;
    unsigned* mmax = (unsigned*)ws; ws += (size_t)NP_ * 4;
    float* denom  = (float*)ws;    ws += (size_t)NP_ * 4;
    float* ebuf   = (float*)ws;    ws += (size_t)E_  * 4;
    float* rn     = (float*)ws;    ws += (size_t)NO_ * 4;

    prep_weights<<<(IN_DIM_ * D_ + 255) / 256, 256, 0, stream>>>(Wm_p, Wm_o, WmTp, WmTo);
    prep_uvec<<<1, 256, 0, stream>>>(W, bW, a_w, u_src, u_dst, consts);
    init_bufs<<<2048, 256, 0, stream>>>(mmax, denom, out1, (long long)NP_ * D_);

    map_gemm<0><<<NP_ / 16, 256, 0, stream>>>(feat_prefix, WmTp, bm_p, u_src, consts, 0,
                                              s_src, nullptr, nullptr);
    map_gemm<1><<<NO_ / 16, 256, 0, stream>>>(feat_other, WmTo, bm_o, u_dst, consts, 1,
                                              s_dst, out2, rn);

    edge_scores<<<(E_ + 255) / 256, 256, 0, stream>>>(src, dst, s_src, s_dst, a_b, ebuf, mmax);
    edge_exp<<<(E_ + 255) / 256, 256, 0, stream>>>(src, mmax, ebuf, denom);
    edge_agg<<<(E_ + 7) / 8, 256, 0, stream>>>(src, dst, ebuf, denom, rn, out2, out1);
    final_norm<<<(NP_ + 7) / 8, 256, 0, stream>>>(out1);
}
